// VectorQuantizerN_84980222919421
// MI455X (gfx1250) — matrix operand order repaired and checked by reading, not compiled, not run on hardware
//
#include <hip/hip_runtime.h>

typedef __attribute__((ext_vector_type(16))) __bf16 v16bf;
typedef __attribute__((ext_vector_type(8)))  float  v8f;
typedef __attribute__((ext_vector_type(4)))  unsigned int u32x4;

static constexpr int NN = 32768;  // rows of z
static constexpr int KK = 8192;   // codebook entries
static constexpr int DD = 512;    // feature dim

// LDS tile: 16 columns x 512 K bf16, padded to 520 shorts per column
// (1040-byte stride => adjacent columns 4 banks apart, conflict-free b128 reads)
static constexpr int COLPAD = 520;
static constexpr int TILE_SH = 16 * COLPAD;   // shorts per buffer

union ABFrag { v16bf v; u32x4 u[2]; };

__device__ __forceinline__ unsigned short f32_to_bf16_rne(float f) {
    unsigned u = __float_as_uint(f);
    unsigned r = u + 0x7FFFu + ((u >> 16) & 1u);
    return (unsigned short)(r >> 16);
}

// ---------------------------------------------------------------------------
// Row L2-normalize codebook -> f32 + bf16 copies in workspace
// ---------------------------------------------------------------------------
__global__ __launch_bounds__(256)
void normalize_codebook_kernel(const float* __restrict__ cb,
                               float* __restrict__ cn_f32,
                               unsigned short* __restrict__ cn_b16) {
    __shared__ float red[8];
    __shared__ float s_inv;
    const int row = blockIdx.x;
    const int t = threadIdx.x;
    const float* p = cb + (size_t)row * DD;
    float x0 = p[t], x1 = p[t + 256];
    float s = x0 * x0 + x1 * x1;
    #pragma unroll
    for (int off = 16; off >= 1; off >>= 1) s += __shfl_down(s, off, 32);
    if ((t & 31) == 0) red[t >> 5] = s;
    __syncthreads();
    if (t == 0) {
        float tot = 0.f;
        #pragma unroll
        for (int i = 0; i < 8; ++i) tot += red[i];
        s_inv = 1.0f / fmaxf(sqrtf(tot), 1e-12f);
    }
    __syncthreads();
    const float inv = s_inv;
    const float y0 = x0 * inv, y1 = x1 * inv;
    const size_t base = (size_t)row * DD;
    cn_f32[base + t] = y0;
    cn_f32[base + t + 256] = y1;
    cn_b16[base + t] = f32_to_bf16_rne(y0);
    cn_b16[base + t + 256] = f32_to_bf16_rne(y1);
}

// ---------------------------------------------------------------------------
// Row L2-normalize z -> bf16 copy in workspace
// ---------------------------------------------------------------------------
__global__ __launch_bounds__(256)
void normalize_z_kernel(const float* __restrict__ z,
                        unsigned short* __restrict__ zn_b16) {
    __shared__ float red[8];
    __shared__ float s_inv;
    const int row = blockIdx.x;
    const int t = threadIdx.x;
    const float* p = z + (size_t)row * DD;
    float x0 = p[t], x1 = p[t + 256];
    float s = x0 * x0 + x1 * x1;
    #pragma unroll
    for (int off = 16; off >= 1; off >>= 1) s += __shfl_down(s, off, 32);
    if ((t & 31) == 0) red[t >> 5] = s;
    __syncthreads();
    if (t == 0) {
        float tot = 0.f;
        #pragma unroll
        for (int i = 0; i < 8; ++i) tot += red[i];
        s_inv = 1.0f / fmaxf(sqrtf(tot), 1e-12f);
    }
    __syncthreads();
    const float inv = s_inv;
    const size_t base = (size_t)row * DD;
    zn_b16[base + t] = f32_to_bf16_rne(x0 * inv);
    zn_b16[base + t + 256] = f32_to_bf16_rne(x1 * inv);
}

// ---------------------------------------------------------------------------
// Fused bf16-WMMA GEMM (zn . cn^T) with streaming row-argmax.
// Block = 8 waves; each wave owns a 16-row strip with A resident in VGPRs.
// B tiles (16 cols x 512 K) are double-buffered in LDS via CDNA5 async
// global->LDS copies (ASYNCcnt), overlapping the next tile's DMA with the
// current tile's 16 chained WMMAs.
// ---------------------------------------------------------------------------
__global__ __launch_bounds__(256)
void argmax_gemm_kernel(const unsigned short* __restrict__ znb,
                        const unsigned short* __restrict__ cnb,
                        int* __restrict__ idx_out) {
    __shared__ unsigned short sB[2 * TILE_SH];   // 2 x 16.6 KB B-tile buffers

    const int tid  = threadIdx.x;
    const int lane = tid & 31;
    const int wave = tid >> 5;
    const int m0 = blockIdx.x * 128 + wave * 16;
    const int l15 = lane & 15;
    const int hi = lane >> 4;   // 0 or 1

    // --- Preload A strip: 16 rows x 512 K, ISA 16-bit A layout ---
    // lanes 0-15: row=m0+l15, K = c*32 + {0..7, 16..23}
    // lanes 16-31: row=m0+l15, K = c*32 + {8..15, 24..31}
    ABFrag a[16];
    const unsigned short* arow = znb + (size_t)(m0 + l15) * DD;
    #pragma unroll
    for (int c = 0; c < 16; ++c) {
        a[c].u[0] = *(const u32x4*)(arow + c * 32 + hi * 8);
        a[c].u[1] = *(const u32x4*)(arow + c * 32 + 16 + hi * 8);
    }

    // Async-stage one 16-col B tile into LDS buffer `buf`.
    // Tile = 1024 16-byte chunks; 256 threads x 4 chunks, each chunk moved by
    // a per-lane global_load_async_to_lds_b128 (32 lanes x 16B per instr).
    auto issue_tile = [&](int n0, int buf) {
        #pragma unroll
        for (int j = 0; j < 4; ++j) {
            const int q   = j * 256 + tid;   // 0..1023
            const int col = q >> 6;          // 0..15
            const int kc  = q & 63;          // 16B chunk within the 512-K row
            const unsigned short* g = cnb + (size_t)(n0 + col) * DD + kc * 8;
            const unsigned short* l = &sB[buf * TILE_SH + col * COLPAD + kc * 8];
            const unsigned ldsOff = (unsigned)(unsigned long long)l;
            asm volatile("global_load_async_to_lds_b128 %0, %1, off"
                         :: "v"(ldsOff), "v"(g) : "memory");
        }
    };

    float bestv[8];
    int   besti[8];
    #pragma unroll
    for (int r = 0; r < 8; ++r) { bestv[r] = -3.4e38f; besti[r] = 0; }

    issue_tile(0, 0);                         // prologue: tile 0 in flight

    for (int i = 0; i < KK / 16; ++i) {
        const int n0 = i * 16;
        if (i + 1 < KK / 16) {
            issue_tile(n0 + 16, (i + 1) & 1); // overlap next tile's DMA
            // 4 newest (next tile) may remain outstanding; current tile done
            asm volatile("s_wait_asynccnt 0x4" ::: "memory");
        } else {
            asm volatile("s_wait_asynccnt 0x0" ::: "memory");
        }
        __syncthreads();                      // tile i visible to all waves

        // B fragment reads from LDS (ISA B layout: lane's column = n0+l15;
        // lanes 0-15 hold K c*32+{0..7,16..23}, lanes 16-31 hold K c*32+{8..15,24..31})
        const unsigned short* bbase =
            sB + (i & 1) * TILE_SH + l15 * COLPAD + hi * 8;
        v8f acc = {};
        #pragma unroll
        for (int c = 0; c < 16; ++c) {
            ABFrag b;
            b.u[0] = *(const u32x4*)(bbase + c * 32);
            b.u[1] = *(const u32x4*)(bbase + c * 32 + 16);
            acc = __builtin_amdgcn_wmma_f32_16x16x32_bf16(
                false, a[c].v, false, b.v, (short)0, acc, false, false);
        }

        // C/D layout: lane n (l15) = column; VGPR r = row r (lanes 0-15)
        // or row 8+r (lanes 16-31).
        const int myidx = n0 + l15;
        #pragma unroll
        for (int r = 0; r < 8; ++r) {
            if (acc[r] > bestv[r]) { bestv[r] = acc[r]; besti[r] = myidx; }
        }
        __syncthreads();                      // all reads of buf i&1 done
    }

    // Reduce argmax across the 16 lanes of each half (xor<=8 stays in-half)
    #pragma unroll
    for (int off = 8; off >= 1; off >>= 1) {
        #pragma unroll
        for (int r = 0; r < 8; ++r) {
            float ov = __shfl_xor(bestv[r], off, 32);
            int   oi = __shfl_xor(besti[r], off, 32);
            if (ov > bestv[r] || (ov == bestv[r] && oi < besti[r])) {
                bestv[r] = ov; besti[r] = oi;
            }
        }
    }
    if (l15 == 0) {
        const int rbase = m0 + hi * 8;  // lane0 -> rows m0..+7, lane16 -> +8..+15
        #pragma unroll
        for (int r = 0; r < 8; ++r) idx_out[rbase + r] = besti[r];
    }
}

// ---------------------------------------------------------------------------
// Gather cn[idx], straight-through output, per-row partial loss (deterministic)
// ---------------------------------------------------------------------------
__global__ __launch_bounds__(256)
void gather_loss_kernel(const float* __restrict__ z,
                        const float* __restrict__ cn_f32,
                        const int* __restrict__ indices,
                        float* __restrict__ zq_out,
                        float* __restrict__ partial) {
    __shared__ float red[8];
    __shared__ float s_inv;
    const int row = blockIdx.x;
    const int t = threadIdx.x;
    const float* p = z + (size_t)row * DD;
    float x0 = p[t], x1 = p[t + 256];
    float s = x0 * x0 + x1 * x1;
    #pragma unroll
    for (int off = 16; off >= 1; off >>= 1) s += __shfl_down(s, off, 32);
    if ((t & 31) == 0) red[t >> 5] = s;
    __syncthreads();
    if (t == 0) {
        float tot = 0.f;
        #pragma unroll
        for (int i = 0; i < 8; ++i) tot += red[i];
        s_inv = 1.0f / fmaxf(sqrtf(tot), 1e-12f);
    }
    __syncthreads();
    const float inv = s_inv;
    const float zn0 = x0 * inv, zn1 = x1 * inv;

    const int idx = indices[row];
    const float* q = cn_f32 + (size_t)idx * DD;
    const float q0 = q[t], q1 = q[t + 256];

    const size_t base = (size_t)row * DD;
    zq_out[base + t]       = zn0 + (q0 - zn0);   // straight-through
    zq_out[base + t + 256] = zn1 + (q1 - zn1);

    const float d0 = q0 - zn0, d1 = q1 - zn1;
    float s2 = d0 * d0 + d1 * d1;
    #pragma unroll
    for (int off = 16; off >= 1; off >>= 1) s2 += __shfl_down(s2, off, 32);
    if ((t & 31) == 0) red[t >> 5] = s2;
    __syncthreads();
    if (t == 0) {
        float tot = 0.f;
        #pragma unroll
        for (int i = 0; i < 8; ++i) tot += red[i];
        partial[row] = tot;
    }
}

__global__ __launch_bounds__(256)
void finalize_loss_kernel(const float* __restrict__ partial,
                          float* __restrict__ loss) {
    __shared__ float red[8];
    const int t = threadIdx.x;
    float s = 0.f;
    for (int i = t; i < NN; i += 256) s += partial[i];
    #pragma unroll
    for (int off = 16; off >= 1; off >>= 1) s += __shfl_down(s, off, 32);
    if ((t & 31) == 0) red[t >> 5] = s;
    __syncthreads();
    if (t == 0) {
        float tot = 0.f;
        #pragma unroll
        for (int i = 0; i < 8; ++i) tot += red[i];
        // loss = BETA*mean + mean = 1.5 * mean((zq - zn)^2)
        *loss = 1.5f * tot / (float)((double)NN * (double)DD);
    }
}

// ---------------------------------------------------------------------------
extern "C" void kernel_launch(void* const* d_in, const int* in_sizes, int n_in,
                              void* d_out, int out_size, void* d_ws, size_t ws_size,
                              hipStream_t stream) {
    const float* z  = (const float*)d_in[0];   // [N, D] f32
    const float* cb = (const float*)d_in[1];   // [K, D] f32

    float* zq    = (float*)d_out;                        // N*D f32
    int*   idxp  = (int*)(zq + (size_t)NN * DD);         // N int32
    float* lossp = (float*)(idxp + NN);                  // 1 f32

    char* w = (char*)d_ws;
    float*          cn_f32  = (float*)w;                                   // 16 MB
    unsigned short* cn_b16  = (unsigned short*)(w + (size_t)KK * DD * 4);  //  8 MB
    unsigned short* zn_b16  = (unsigned short*)(w + (size_t)KK * DD * 6);  // 32 MB
    float*          partial = (float*)(w + (size_t)KK * DD * 6 + (size_t)NN * DD * 2);

    normalize_codebook_kernel<<<KK, 256, 0, stream>>>(cb, cn_f32, cn_b16);
    normalize_z_kernel<<<NN, 256, 0, stream>>>(z, zn_b16);
    argmax_gemm_kernel<<<NN / 128, 256, 0, stream>>>(zn_b16, cn_b16, idxp);
    gather_loss_kernel<<<NN, 256, 0, stream>>>(z, cn_f32, idxp, zq, partial);
    finalize_loss_kernel<<<1, 256, 0, stream>>>(partial, lossp);
}
